// ProteinGNN_43645457662487
// MI455X (gfx1250) — matrix operand order repaired and checked by reading, not compiled, not run on hardware
//
#include <hip/hip_runtime.h>
#include <hip/hip_bf16.h>
#include <math.h>

// ---------------- problem constants ----------------
#define N_AA    300000
#define N_PROT  16384
#define C_DIM   128
#define VOCAB   5000
#define AA_IN   64
#define PROT_IN 256
#define E_A2P   300000

// ---------------- plain vector types (safe in unions) ----------------
typedef __attribute__((ext_vector_type(16))) __bf16    v16bf;
typedef __attribute__((ext_vector_type(8)))  float     v8f;
typedef __attribute__((ext_vector_type(4)))  float     f32x4;
typedef __attribute__((ext_vector_type(4)))  unsigned  u32x4;

union BF16x16 {
    v16bf v;
    u32x4 q[2];
    unsigned short s[16];
};

__device__ __forceinline__ unsigned short f2bf(float f) {
    unsigned u = __float_as_uint(f);
    u += 0x7FFFu + ((u >> 16) & 1u);          // round-to-nearest-even
    return (unsigned short)(u >> 16);
}

// ============================================================
// Weight pre-conversion: W[K,N] f32 -> Wb[K,Npad] bf16 (zero-padded)
// ============================================================
__global__ void convert_w_bf16(const float* __restrict__ W,
                               unsigned short* __restrict__ Wb,
                               int K, int N, int Npad) {
    const size_t i = blockIdx.x * (size_t)blockDim.x + threadIdx.x;
    if (i >= (size_t)K * Npad) return;
    const int k = (int)(i / Npad), c = (int)(i % Npad);
    Wb[i] = (c < N) ? f2bf(W[(size_t)k * N + c]) : (unsigned short)0;
}

// ============================================================
// GEMM: out[M,N] = act(A[M,K] @ W[K,N] + bias[N])
// A: f32 row-major, Wb: bf16 row-major [K, Npad], Npad % 128 == 0.
// One wave per 16x128 tile; bf16 WMMA 16x16x32, f32 accumulate.
// M % 16 == 0, K % 32 == 0. act: 0 = none, 1 = sigmoid.
// ============================================================
__global__ __launch_bounds__(32)
void gemm_wmma_bf16(const float* __restrict__ A,
                    const unsigned short* __restrict__ Wb,
                    const float* __restrict__ bias, float* __restrict__ out,
                    int M, int K, int N, int Npad, int act) {
    const int row0 = blockIdx.x * 16;
    const int n0   = blockIdx.y * 128;
    const int lane = threadIdx.x;          // 0..31 (wave32)
    const int m    = lane & 15;
    const int hk   = lane >> 4;            // K-half select (ISA A layout)

    v8f acc[8] = {};                       // 8 N-tiles of 16

    for (int k0 = 0; k0 < K; k0 += 32) {
        // ---- A fragment: lane holds row (row0+m); two contiguous 8-float runs
        const float* ap = A + (size_t)(row0 + m) * K + k0 + 8 * hk;
        const f32x4 a0 = ((const f32x4*)ap)[0];
        const f32x4 a1 = ((const f32x4*)ap)[1];
        const f32x4 a2 = ((const f32x4*)(ap + 16))[0];
        const f32x4 a3 = ((const f32x4*)(ap + 16))[1];
        BF16x16 a;
#pragma unroll
        for (int i = 0; i < 4; ++i) {
            a.s[i]      = f2bf(a0[i]);
            a.s[4 + i]  = f2bf(a1[i]);
            a.s[8 + i]  = f2bf(a2[i]);
            a.s[12 + i] = f2bf(a3[i]);
        }

        // ---- B fragments: lane holds N-col (m); K rows k0+8*hk+{0..7,16..23}
        const unsigned short* wcol = Wb + (size_t)(k0 + 8 * hk) * Npad + n0 + m;
#pragma unroll
        for (int t = 0; t < 8; ++t) {
            BF16x16 b;
#pragma unroll
            for (int i = 0; i < 8; ++i) {
                b.s[i]     = wcol[(size_t)i * Npad + t * 16];
                b.s[8 + i] = wcol[(size_t)(16 + i) * Npad + t * 16];
            }
            acc[t] = __builtin_amdgcn_wmma_f32_16x16x32_bf16(
                false, a.v, false, b.v, (short)0, acc[t], false, false);
        }
    }

    // ---- store: D layout -> lane (l&15) = N col, vgpr e -> row e+8*hk
#pragma unroll
    for (int t = 0; t < 8; ++t) {
        const int col = n0 + t * 16 + m;
        if (col < N) {
            const float bv = bias ? bias[col] : 0.0f;
#pragma unroll
            for (int e = 0; e < 8; ++e) {
                const int row = row0 + e + 8 * hk;
                float v = acc[t][e] + bv;
                if (act == 1) v = 1.0f / (1.0f + __expf(-v));
                out[(size_t)row * N + col] = v;
            }
        }
    }
}

// ============================================================
// Fused (optional bias) + PReLU + LayerNorm, one block per row.
// ============================================================
__global__ void prelu_ln_kernel(const float* __restrict__ x,
                                const float* __restrict__ bias,
                                const float* __restrict__ alpha, int alpha_vec,
                                const float* __restrict__ g,
                                const float* __restrict__ b,
                                float* __restrict__ out, int N) {
    const int r = blockIdx.x, c = threadIdx.x;
    float v = x[(size_t)r * N + c];
    if (bias) v += bias[c];
    const float a = alpha_vec ? alpha[c] : alpha[0];
    v = (v >= 0.0f) ? v : a * v;

    __shared__ float ssum[8], ssum2[8];
    float s1 = v, s2 = v * v;
#pragma unroll
    for (int off = 16; off > 0; off >>= 1) {
        s1 += __shfl_xor(s1, off, 32);
        s2 += __shfl_xor(s2, off, 32);
    }
    const int wid = c >> 5;
    if ((c & 31) == 0) { ssum[wid] = s1; ssum2[wid] = s2; }
    __syncthreads();
    if (c == 0) {
        const int nw = N >> 5;
        float t1 = 0.0f, t2 = 0.0f;
        for (int i = 0; i < nw; ++i) { t1 += ssum[i]; t2 += ssum2[i]; }
        ssum[0] = t1; ssum2[0] = t2;
    }
    __syncthreads();
    const float inv = 1.0f / (float)N;
    const float mu  = ssum[0] * inv;
    const float var = ssum2[0] * inv - mu * mu;
    out[(size_t)r * N + c] = g[c] * (v - mu) * rsqrtf(var + 1e-5f) + b[c];
}

// ---------------- order-preserving float<->uint for atomic max ----------------
__device__ __forceinline__ unsigned f_enc(float f) {
    unsigned u = __float_as_uint(f);
    return (u & 0x80000000u) ? ~u : (u | 0x80000000u);
}
__device__ __forceinline__ float f_dec(unsigned u) {
    unsigned v = (u & 0x80000000u) ? (u & 0x7FFFFFFFu) : ~u;
    return __uint_as_float(v);
}

// ============================================================
// GATv2 edge kernels (one wave per edge; 4 channels/lane)
// ============================================================
__global__ __launch_bounds__(256)
void gat_edge_logits(const float* __restrict__ xl, const float* __restrict__ xr,
                     const int* __restrict__ src, const int* __restrict__ dst,
                     const float* __restrict__ att,
                     float* __restrict__ ebuf, unsigned* __restrict__ menc, int E) {
    const int w = (int)(blockIdx.x * 8u + (threadIdx.x >> 5));
    if (w >= E) return;
    const int lane = threadIdx.x & 31;
    const int si = src[w], di = dst[w];
    const float* pl = xl + (size_t)si * C_DIM;
    const float* pr = xr + (size_t)di * C_DIM;
    float acc = 0.0f;
#pragma unroll
    for (int c = lane; c < C_DIM; c += 32) {
        float t = pl[c] + pr[c];
        t = (t > 0.0f) ? t : 0.2f * t;        // leaky_relu(0.2)
        acc += att[c] * t;
    }
#pragma unroll
    for (int off = 16; off > 0; off >>= 1) acc += __shfl_xor(acc, off, 32);
    if (lane == 0) {
        ebuf[w] = acc;
        atomicMax(&menc[di], f_enc(acc));
    }
}

__global__ void gat_edge_expsum(const int* __restrict__ dst,
                                float* __restrict__ ebuf,
                                const unsigned* __restrict__ menc,
                                float* __restrict__ sbuf, int E) {
    const int i = blockIdx.x * blockDim.x + threadIdx.x;
    if (i >= E) return;
    const int di = dst[i];
    const float ex = __expf(ebuf[i] - f_dec(menc[di]));
    ebuf[i] = ex;
    atomicAdd(&sbuf[di], ex);
}

__global__ __launch_bounds__(256)
void gat_edge_scatter(const float* __restrict__ xl,
                      const int* __restrict__ src, const int* __restrict__ dst,
                      const float* __restrict__ ebuf, const float* __restrict__ sbuf,
                      float* __restrict__ acc_out, int E) {
    const int w = (int)(blockIdx.x * 8u + (threadIdx.x >> 5));
    if (w >= E) return;
    const int lane = threadIdx.x & 31;
    const int si = src[w], di = dst[w];
    const float alpha = ebuf[w] / (sbuf[di] + 1e-16f);
    const float* pl = xl + (size_t)si * C_DIM;
    float* po = acc_out + (size_t)di * C_DIM;
#pragma unroll
    for (int c = lane; c < C_DIM; c += 32) atomicAdd(&po[c], alpha * pl[c]);
}

// ---------------- helpers ----------------
__global__ void fill_f32(float* p, float v, size_t n) {
    const size_t i = blockIdx.x * (size_t)blockDim.x + threadIdx.x;
    if (i < n) p[i] = v;
}

__global__ void concat_kernel(const float* __restrict__ a,
                              const float* __restrict__ b,
                              float* __restrict__ out, int M) {
    const size_t i = blockIdx.x * (size_t)blockDim.x + threadIdx.x;
    if (i >= (size_t)M * 256) return;
    const int r = (int)(i >> 8), c = (int)(i & 255);
    out[i] = (c < 128) ? a[(size_t)r * 128 + c] : b[(size_t)r * 128 + (c - 128)];
}

// ============================================================
// host-side orchestration
// ============================================================
static inline void run_conv_w(const float* W, unsigned short* Wb, int K, int N,
                              int Npad, hipStream_t s) {
    const size_t n = (size_t)K * Npad;
    convert_w_bf16<<<(unsigned)((n + 255) / 256), 256, 0, s>>>(W, Wb, K, N, Npad);
}
static inline void run_gemm(const float* A, const unsigned short* Wb,
                            const float* bias, float* out,
                            int M, int K, int N, int Npad, int act,
                            hipStream_t s) {
    dim3 grid(M / 16, Npad / 128);
    gemm_wmma_bf16<<<grid, 32, 0, s>>>(A, Wb, bias, out, M, K, N, Npad, act);
}
static inline void run_fill(float* p, float v, size_t n, hipStream_t s) {
    fill_f32<<<(unsigned)((n + 255) / 256), 256, 0, s>>>(p, v, n);
}

extern "C" void kernel_launch(void* const* d_in, const int* in_sizes, int n_in,
                              void* d_out, int out_size, void* d_ws, size_t ws_size,
                              hipStream_t stream) {
    (void)in_sizes; (void)n_in; (void)out_size; (void)ws_size;
    #define PF(i) ((const float*)d_in[(i)])

    const float* x_aa = PF(0);
    const float* x_pr = PF(1);
    const int* ei     = (const int*)d_in[2];        // ei_a2p [2, E]
    const int* e_src  = ei;                         // aa indices
    const int* e_dst  = ei + E_A2P;                 // protein indices

    const int VPAD = 5120;                          // VOCAB rounded to 128

    // ---- workspace carve-up (floats) ----
    float* ws = (float*)d_ws;
    size_t o = 0;
    float* h_aa  = ws + o; o += (size_t)N_AA   * C_DIM;   // 38.4M
    float* h_pr  = ws + o; o += (size_t)N_PROT * C_DIM;
    float* xl    = ws + o; o += (size_t)N_AA   * C_DIM;   // 38.4M (also GEMM tmp)
    float* xr    = ws + o; o += (size_t)N_PROT * C_DIM;   // (also GEMM tmp)
    float* gacc  = ws + o; o += (size_t)N_PROT * C_DIM;
    float* g1    = ws + o; o += (size_t)N_PROT * C_DIM;
    float* g2    = ws + o; o += (size_t)N_PROT * C_DIM;
    float* xp    = ws + o; o += (size_t)N_PROT * 2 * C_DIM;
    float* xp2   = ws + o; o += (size_t)N_PROT * C_DIM;
    float* ebuf  = ws + o; o += (size_t)E_A2P;
    unsigned* menc = (unsigned*)(ws + o); o += (size_t)N_PROT;
    float* sbuf  = ws + o; o += (size_t)N_PROT;
    // bf16 weight buffers (2 shorts per float slot)
    unsigned short* wb = (unsigned short*)(ws + o);
    size_t wo = 0;
    unsigned short* wb_in_aa = wb + wo; wo += (size_t)AA_IN   * C_DIM;
    unsigned short* wb_in_pr = wb + wo; wo += (size_t)PROT_IN * C_DIM;
    unsigned short* wb_l1    = wb + wo; wo += (size_t)C_DIM   * C_DIM;
    unsigned short* wb_r1    = wb + wo; wo += (size_t)C_DIM   * C_DIM;
    unsigned short* wb_l2    = wb + wo; wo += (size_t)C_DIM   * C_DIM;
    unsigned short* wb_r2    = wb + wo; wo += (size_t)C_DIM   * C_DIM;
    unsigned short* wb_post  = wb + wo; wo += (size_t)(2 * C_DIM) * C_DIM;
    unsigned short* wb_out   = wb + wo; wo += (size_t)C_DIM * VPAD;

    const unsigned edgeW = (E_A2P + 7) / 8;         // wave-per-edge grids
    const unsigned edgeT = (E_A2P + 255) / 256;

    // ---- pre-convert all weights to padded bf16 (tiny, L2-resident) ----
    run_conv_w(PF(5),  wb_in_aa, AA_IN,   C_DIM, C_DIM, stream);
    run_conv_w(PF(7),  wb_in_pr, PROT_IN, C_DIM, C_DIM, stream);
    run_conv_w(PF(15), wb_l1,    C_DIM,   C_DIM, C_DIM, stream);
    run_conv_w(PF(17), wb_r1,    C_DIM,   C_DIM, C_DIM, stream);
    run_conv_w(PF(36), wb_l2,    C_DIM,   C_DIM, C_DIM, stream);
    run_conv_w(PF(38), wb_r2,    C_DIM,   C_DIM, C_DIM, stream);
    run_conv_w(PF(60), wb_post,  2*C_DIM, C_DIM, C_DIM, stream);
    run_conv_w(PF(65), wb_out,   C_DIM,   VOCAB, VPAD,  stream);

    // ---- input embeddings: LN(PReLU(x @ W + b)) ----
    run_gemm(x_aa, wb_in_aa, PF(6), xl, N_AA, AA_IN, C_DIM, C_DIM, 0, stream);
    prelu_ln_kernel<<<N_AA, C_DIM, 0, stream>>>(xl, nullptr, PF(9), 1, PF(11), PF(12), h_aa, C_DIM);
    run_gemm(x_pr, wb_in_pr, PF(8), xr, N_PROT, PROT_IN, C_DIM, C_DIM, 0, stream);
    prelu_ln_kernel<<<N_PROT, C_DIM, 0, stream>>>(xr, nullptr, PF(10), 1, PF(13), PF(14), h_pr, C_DIM);

    // ---- GATv2 a2p, layer 1 (p2a/a2a outputs are dead in the reference) ----
    run_gemm(h_aa, wb_l1, PF(16), xl, N_AA,   C_DIM, C_DIM, C_DIM, 0, stream);
    run_gemm(h_pr, wb_r1, PF(18), xr, N_PROT, C_DIM, C_DIM, C_DIM, 0, stream);
    run_fill((float*)menc, 0.0f, N_PROT, stream);   // 0 sorts below enc(-inf)
    run_fill(sbuf, 0.0f, N_PROT, stream);
    run_fill(gacc, 0.0f, (size_t)N_PROT * C_DIM, stream);
    gat_edge_logits <<<edgeW, 256, 0, stream>>>(xl, xr, e_src, e_dst, PF(19), ebuf, menc, E_A2P);
    gat_edge_expsum <<<edgeT, 256, 0, stream>>>(e_dst, ebuf, menc, sbuf, E_A2P);
    gat_edge_scatter<<<edgeW, 256, 0, stream>>>(xl, e_src, e_dst, ebuf, sbuf, gacc, E_A2P);
    prelu_ln_kernel<<<N_PROT, C_DIM, 0, stream>>>(gacc, PF(20), PF(33), 0, PF(34), PF(35), g1, C_DIM);

    // ---- GATv2 a2p, layer 2 (src feats = h_aa via skip connection) ----
    run_gemm(h_aa, wb_l2, PF(37), xl, N_AA,   C_DIM, C_DIM, C_DIM, 0, stream);
    run_gemm(g1,   wb_r2, PF(39), xr, N_PROT, C_DIM, C_DIM, C_DIM, 0, stream);
    run_fill((float*)menc, 0.0f, N_PROT, stream);
    run_fill(sbuf, 0.0f, N_PROT, stream);
    run_fill(gacc, 0.0f, (size_t)N_PROT * C_DIM, stream);
    gat_edge_logits <<<edgeW, 256, 0, stream>>>(xl, xr, e_src, e_dst, PF(40), ebuf, menc, E_A2P);
    gat_edge_expsum <<<edgeT, 256, 0, stream>>>(e_dst, ebuf, menc, sbuf, E_A2P);
    gat_edge_scatter<<<edgeW, 256, 0, stream>>>(xl, e_src, e_dst, ebuf, sbuf, gacc, E_A2P);
    prelu_ln_kernel<<<N_PROT, C_DIM, 0, stream>>>(gacc, PF(41), PF(54), 0, PF(55), PF(56), g2, C_DIM);

    // ---- head: concat -> LN -> lin_post -> LN -> lin_out -> sigmoid ----
    concat_kernel<<<(N_PROT * 256 + 255) / 256, 256, 0, stream>>>(h_pr, g2, xp, N_PROT);
    prelu_ln_kernel<<<N_PROT, 2 * C_DIM, 0, stream>>>(xp, nullptr, PF(57), 0, PF(58), PF(59), xp, 2 * C_DIM);
    run_gemm(xp, wb_post, PF(61), xp2, N_PROT, 2 * C_DIM, C_DIM, C_DIM, 0, stream);
    prelu_ln_kernel<<<N_PROT, C_DIM, 0, stream>>>(xp2, nullptr, PF(62), 0, PF(63), PF(64), xp2, C_DIM);
    run_gemm(xp2, wb_out, PF(66), (float*)d_out, N_PROT, C_DIM, VOCAB, VPAD, 1, stream);

    #undef PF
}
